// GIN0_16295105921239
// MI455X (gfx1250) — matrix operand order repaired and checked by reading, not compiled, not run on hardware
//
#include <hip/hip_runtime.h>
#include <hip/hip_bf16.h>

// ---------------------------------------------------------------------------
// GIN forward for MI455X (gfx1250, wave32, WMMA).
// N=100000 nodes, E=1.6M edges, H=64, L=3 layers, G=512 graphs.
// Memory-bound: node features = 25.6MB (L2-resident). GEMMs done with
// v_wmma_f32_16x16x32_f16; weights pre-converted once to fragment layout.
// ---------------------------------------------------------------------------

typedef __attribute__((ext_vector_type(16))) _Float16 v16h;
typedef __attribute__((ext_vector_type(8)))  _Float16 v8h;
typedef __attribute__((ext_vector_type(8)))  float    v8f;

#define GIN_N 100000
#define GIN_E 1600000
#define GIN_H 64
#define GIN_L 3
#define GIN_G 512

// ---- workspace layout (bytes) ----
#define NH_BYTES   (GIN_N * GIN_H * 4)         // 25,600,000
#define OFF_XBUF0  ((size_t)0)
#define OFF_XBUF1  ((size_t)NH_BYTES)
#define OFF_AGG    ((size_t)NH_BYTES * 2)
#define OFF_WFRAG  ((size_t)NH_BYTES * 3)      // 32768 halves = 65536 B
#define OFF_AC     (OFF_WFRAG + 65536)         // 768 floats = 3072 B
#define OFF_POOL   (OFF_AC + 3072)             // 32768 floats = 131072 B

// ---------------------------------------------------------------------------
// A-fragment loader for v_wmma_f32_16x16x32_f16.
// Per ISA 16-bit A layout: lane (lo = lane&15 selects row M, hi = lane>>4)
// holds two contiguous runs of 8 K-values:  K = kc*32 + hi*8 + {0..7} and
// K = kc*32 + hi*8 + 16 + {0..7}.  p = row + kc*32 + hi*8.
// ---------------------------------------------------------------------------
__device__ __forceinline__ v16h load_a_f32(const float* __restrict__ p) {
  const float4 r0 = *(const float4*)(p);
  const float4 r1 = *(const float4*)(p + 4);
  const float4 r2 = *(const float4*)(p + 16);
  const float4 r3 = *(const float4*)(p + 20);
  v16h a;
  a[0]  = (_Float16)r0.x; a[1]  = (_Float16)r0.y; a[2]  = (_Float16)r0.z; a[3]  = (_Float16)r0.w;
  a[4]  = (_Float16)r1.x; a[5]  = (_Float16)r1.y; a[6]  = (_Float16)r1.z; a[7]  = (_Float16)r1.w;
  a[8]  = (_Float16)r2.x; a[9]  = (_Float16)r2.y; a[10] = (_Float16)r2.z; a[11] = (_Float16)r2.w;
  a[12] = (_Float16)r3.x; a[13] = (_Float16)r3.y; a[14] = (_Float16)r3.z; a[15] = (_Float16)r3.w;
  return a;
}

__device__ __forceinline__ v16h cat8(v8h x, v8h y) {
  v16h a;
#pragma unroll
  for (int i = 0; i < 8; ++i) { a[i] = x[i]; a[i + 8] = y[i]; }
  return a;
}

// ---------------------------------------------------------------------------
// prep: weights fp32 -> f16 B-fragment layout; fold eval-BN into (a, c).
// B-frag mapping (16x16x32, B is 32x16 KxN): element e of lane L holds
// B[kc*32 + (e&7) + 8*(L>>4) + 16*(e>>3)][nt*16 + (L&15)].
// wfrag layout: [w(0..5)]*4096 + ((kc*4+nt)*32 + lane)*16 + e ; linW at 24576
// with fidx = kc*8+nt. ac layout: (l*2+j)*128 + {0..63: a, 64..127: c}.
// ---------------------------------------------------------------------------
__global__ void gin_prep_kernel(
    const float* __restrict__ W1, const float* __restrict__ W2,
    const float* __restrict__ linW,
    const float* __restrict__ b1, const float* __restrict__ g1,
    const float* __restrict__ bt1, const float* __restrict__ rm1,
    const float* __restrict__ rv1,
    const float* __restrict__ b2, const float* __restrict__ g2,
    const float* __restrict__ bt2, const float* __restrict__ rm2,
    const float* __restrict__ rv2,
    _Float16* __restrict__ wfrag, float* __restrict__ ac) {
  int tid = blockIdx.x * blockDim.x + threadIdx.x;
  if (tid < 24576) {                       // 6 GIN weight matrices, 64x64 each
    int w = tid >> 12;
    int r = tid & 4095;
    int e = r & 15, lane = (r >> 4) & 31, fidx = r >> 9;   // fidx = kc*4+nt
    int kc = fidx >> 2, nt = fidx & 3;
    int k = kc * 32 + (e & 7) + 8 * (lane >> 4) + 16 * (e >> 3);
    int n = nt * 16 + (lane & 15);
    int l = w >> 1;
    const float* W = (w & 1) ? W2 : W1;
    wfrag[tid] = (_Float16)W[l * 4096 + k * 64 + n];
  } else if (tid < 32768) {                // linW 64x128
    int r = tid - 24576;
    int e = r & 15, lane = (r >> 4) & 31, fidx = r >> 9;   // fidx = kc*8+nt
    int kc = fidx >> 3, nt = fidx & 7;
    int k = kc * 32 + (e & 7) + 8 * (lane >> 4) + 16 * (e >> 3);
    int n = nt * 16 + (lane & 15);
    wfrag[tid] = (_Float16)linW[k * 128 + n];
  } else if (tid < 32768 + 384) {          // folded BN constants
    int r = tid - 32768;
    int lj = r >> 6, n = r & 63;
    int l = lj >> 1, j = lj & 1;
    const float* g  = j ? g2  : g1;
    const float* bt = j ? bt2 : bt1;
    const float* rm = j ? rm2 : rm1;
    const float* rv = j ? rv2 : rv1;
    const float* bb = j ? b2  : b1;
    float a = g[l * 64 + n] * rsqrtf(rv[l * 64 + n] + 1e-5f);
    float c = (bb[l * 64 + n] - rm[l * 64 + n]) * a + bt[l * 64 + n];
    ac[lj * 128 + n]      = a;
    ac[lj * 128 + 64 + n] = c;
  }
}

// ---------------------------------------------------------------------------
// agg <- x  (GIN eps=0: h = x + sum of neighbors, so seed agg with x)
// ---------------------------------------------------------------------------
__global__ void gin_copy4_kernel(const float4* __restrict__ src,
                                 float4* __restrict__ dst, int n4) {
  for (int i = blockIdx.x * blockDim.x + threadIdx.x; i < n4;
       i += gridDim.x * blockDim.x)
    dst[i] = src[i];
}

__global__ void gin_zero4_kernel(float4* __restrict__ dst, int n4) {
  int i = blockIdx.x * blockDim.x + threadIdx.x;
  if (i < n4) dst[i] = make_float4(0.f, 0.f, 0.f, 0.f);
}

// ---------------------------------------------------------------------------
// scatter: agg[dst[e]] += x[src[e]]  (thread = edge * 16 feature-quads)
// ---------------------------------------------------------------------------
__global__ void gin_scatter_kernel(const float* __restrict__ x,
                                   const int* __restrict__ src,
                                   const int* __restrict__ dst,
                                   float* __restrict__ agg) {
  int tid = blockIdx.x * blockDim.x + threadIdx.x;
  if (tid >= GIN_E * 16) return;
  int e = tid >> 4;
  int f = (tid & 15) << 2;
  int s = src[e], d = dst[e];
  float4 v = *(const float4*)(x + (size_t)s * GIN_H + f);
  float* p = agg + (size_t)d * GIN_H + f;
  atomicAdd(p + 0, v.x); atomicAdd(p + 1, v.y);
  atomicAdd(p + 2, v.z); atomicAdd(p + 3, v.w);
}

// ---------------------------------------------------------------------------
// pool: pooled[batch[n]] += x[n]
// ---------------------------------------------------------------------------
__global__ void gin_pool_kernel(const float* __restrict__ x,
                                const int* __restrict__ batch,
                                float* __restrict__ pooled) {
  int tid = blockIdx.x * blockDim.x + threadIdx.x;
  if (tid >= GIN_N * 16) return;
  int n = tid >> 4;
  int f = (tid & 15) << 2;
  int g = batch[n];
  float4 v = *(const float4*)(x + (size_t)n * GIN_H + f);
  float* p = pooled + (size_t)g * GIN_H + f;
  atomicAdd(p + 0, v.x); atomicAdd(p + 1, v.y);
  atomicAdd(p + 2, v.z); atomicAdd(p + 3, v.w);
}

// ---------------------------------------------------------------------------
// Fused MLP: one wave owns a 16x64 row strip.
//   h1 = relu(a1*(A@W1) + c1) ; out = relu(a2*(h1@W2) + c2)
// GEMM1 A-frags straight from global fp32; intermediate staged per-wave in
// LDS as f16 (D-layout -> A-layout transpose across lanes), s_wait_dscnt,
// GEMM2, fp32 store. 16 v_wmma per wave. No block barriers.
// ---------------------------------------------------------------------------
__global__ __launch_bounds__(256) void gin_mlp_kernel(
    const float* __restrict__ agg, const _Float16* __restrict__ wf1,
    const _Float16* __restrict__ wf2, const float* __restrict__ ac,
    float* __restrict__ xout) {
  __shared__ __align__(16) _Float16 lds[8 * 16 * GIN_H];  // 2KB per wave
  const int lane = threadIdx.x & 31;
  const int wave = threadIdx.x >> 5;
  const int strip = blockIdx.x * 8 + wave;
  if (strip * 16 >= GIN_N) return;
  _Float16* hbuf = lds + wave * (16 * GIN_H);
  const int hi = lane >> 4, lo = lane & 15;
  const int m = strip * 16 + lo;

  const float* a1p = ac;        const float* c1p = ac + 64;
  const float* a2p = ac + 128;  const float* c2p = ac + 192;

  // ---- GEMM1 A fragments (fp32 -> f16) ----
  const float* arow = agg + (size_t)m * GIN_H + hi * 8;
  v16h A0 = load_a_f32(arow);        // K chunk 0
  v16h A1 = load_a_f32(arow + 32);   // K chunk 1

  // ---- GEMM1 + BN/ReLU epilogue -> LDS (f16, row major 16x64) ----
#pragma unroll
  for (int nt = 0; nt < 4; ++nt) {
    v8f acc = {};
    v16h B0 = *(const v16h*)(wf1 + ((0 * 4 + nt) * 32 + lane) * 16);
    acc = __builtin_amdgcn_wmma_f32_16x16x32_f16(false, A0, false, B0,
                                                 (short)0, acc, false, false);
    v16h B1 = *(const v16h*)(wf1 + ((1 * 4 + nt) * 32 + lane) * 16);
    acc = __builtin_amdgcn_wmma_f32_16x16x32_f16(false, A1, false, B1,
                                                 (short)0, acc, false, false);
    const float av = a1p[nt * 16 + lo], cv = c1p[nt * 16 + lo];
#pragma unroll
    for (int r = 0; r < 8; ++r) {
      float h = fmaxf(acc[r] * av + cv, 0.0f);
      hbuf[(r + 8 * hi) * GIN_H + nt * 16 + lo] = (_Float16)h;
    }
  }

  asm volatile("s_wait_dscnt 0" ::: "memory");  // intra-wave LDS RAW

  // ---- rebuild A fragments for GEMM2 from LDS ----
  const _Float16* hrow = hbuf + lo * GIN_H + hi * 8;
  v16h H0 = cat8(*(const v8h*)(hrow),      *(const v8h*)(hrow + 16));
  v16h H1 = cat8(*(const v8h*)(hrow + 32), *(const v8h*)(hrow + 48));

  // ---- GEMM2 + BN/ReLU epilogue -> global fp32 ----
#pragma unroll
  for (int nt = 0; nt < 4; ++nt) {
    v8f acc = {};
    v16h B0 = *(const v16h*)(wf2 + ((0 * 4 + nt) * 32 + lane) * 16);
    acc = __builtin_amdgcn_wmma_f32_16x16x32_f16(false, H0, false, B0,
                                                 (short)0, acc, false, false);
    v16h B1 = *(const v16h*)(wf2 + ((1 * 4 + nt) * 32 + lane) * 16);
    acc = __builtin_amdgcn_wmma_f32_16x16x32_f16(false, H1, false, B1,
                                                 (short)0, acc, false, false);
    const float av = a2p[nt * 16 + lo], cv = c2p[nt * 16 + lo];
#pragma unroll
    for (int r = 0; r < 8; ++r) {
      float h = fmaxf(acc[r] * av + cv, 0.0f);
      xout[(size_t)(strip * 16 + r + 8 * hi) * GIN_H + nt * 16 + lo] = h;
    }
  }
}

// ---------------------------------------------------------------------------
// Final linear: [512,64] @ [64,128] + linb -> d_out. wave = (mt, nt) tile.
// ---------------------------------------------------------------------------
__global__ __launch_bounds__(256) void gin_final_kernel(
    const float* __restrict__ pooled, const _Float16* __restrict__ wflin,
    const float* __restrict__ linb, float* __restrict__ out) {
  const int lane = threadIdx.x & 31;
  const int nt = threadIdx.x >> 5;   // 0..7 (128/16 col tiles)
  const int mt = blockIdx.x;         // 0..31 (512/16 row tiles)
  const int hi = lane >> 4, lo = lane & 15;
  const int m = mt * 16 + lo;

  const float* arow = pooled + (size_t)m * GIN_H + hi * 8;
  v16h A0 = load_a_f32(arow);
  v16h A1 = load_a_f32(arow + 32);

  v8f acc = {};
  v16h B0 = *(const v16h*)(wflin + ((0 * 8 + nt) * 32 + lane) * 16);
  acc = __builtin_amdgcn_wmma_f32_16x16x32_f16(false, A0, false, B0,
                                               (short)0, acc, false, false);
  v16h B1 = *(const v16h*)(wflin + ((1 * 8 + nt) * 32 + lane) * 16);
  acc = __builtin_amdgcn_wmma_f32_16x16x32_f16(false, A1, false, B1,
                                               (short)0, acc, false, false);

  const float lb = linb[nt * 16 + lo];
#pragma unroll
  for (int r = 0; r < 8; ++r)
    out[(size_t)(mt * 16 + r + 8 * hi) * 128 + nt * 16 + lo] = acc[r] + lb;
}

// ---------------------------------------------------------------------------
extern "C" void kernel_launch(void* const* d_in, const int* in_sizes, int n_in,
                              void* d_out, int out_size, void* d_ws,
                              size_t ws_size, hipStream_t stream) {
  const float* x      = (const float*)d_in[0];
  const int*   ei     = (const int*)d_in[1];
  const int*   batch  = (const int*)d_in[2];
  const float* W1     = (const float*)d_in[3];
  const float* b1     = (const float*)d_in[4];
  const float* g1     = (const float*)d_in[5];
  const float* bt1    = (const float*)d_in[6];
  const float* rm1    = (const float*)d_in[7];
  const float* rv1    = (const float*)d_in[8];
  const float* W2     = (const float*)d_in[9];
  const float* b2     = (const float*)d_in[10];
  const float* g2     = (const float*)d_in[11];
  const float* bt2    = (const float*)d_in[12];
  const float* rm2    = (const float*)d_in[13];
  const float* rv2    = (const float*)d_in[14];
  const float* linW   = (const float*)d_in[15];
  const float* linb   = (const float*)d_in[16];
  const int* src = ei;
  const int* dst = ei + GIN_E;

  char* ws = (char*)d_ws;
  float*     xbuf0  = (float*)(ws + OFF_XBUF0);
  float*     xbuf1  = (float*)(ws + OFF_XBUF1);
  float*     agg    = (float*)(ws + OFF_AGG);
  _Float16*  wfrag  = (_Float16*)(ws + OFF_WFRAG);
  float*     ac     = (float*)(ws + OFF_AC);
  float*     pooled = (float*)(ws + OFF_POOL);

  // 1. weight fragments + folded BN constants
  gin_prep_kernel<<<130, 256, 0, stream>>>(W1, W2, linW, b1, g1, bt1, rm1, rv1,
                                           b2, g2, bt2, rm2, rv2, wfrag, ac);

  const int nh4 = GIN_N * GIN_H / 4;            // 1,600,000 float4
  const int scatter_blocks = (GIN_E * 16) / 256; // 100000
  const float* xcur = x;
  float* xbufs[2] = {xbuf0, xbuf1};

  for (int l = 0; l < GIN_L; ++l) {
    float* xnext = xbufs[l & 1] == xcur ? xbufs[(l + 1) & 1] : xbufs[l & 1];
    // agg = x ; agg += neighbor sums
    gin_copy4_kernel<<<2048, 256, 0, stream>>>((const float4*)xcur,
                                               (float4*)agg, nh4);
    gin_scatter_kernel<<<scatter_blocks, 256, 0, stream>>>(xcur, src, dst, agg);
    // fused 2-layer MLP with WMMA
    gin_mlp_kernel<<<(GIN_N / 16 + 7) / 8, 256, 0, stream>>>(
        agg, wfrag + (size_t)(l * 2) * 4096, wfrag + (size_t)(l * 2 + 1) * 4096,
        ac + (size_t)l * 256, xnext);
    xcur = xnext;
  }

  // 2. graph pooling
  gin_zero4_kernel<<<(GIN_G * GIN_H / 4 + 255) / 256, 256, 0, stream>>>(
      (float4*)pooled, GIN_G * GIN_H / 4);
  gin_pool_kernel<<<(GIN_N * 16 + 255) / 256, 256, 0, stream>>>(xcur, batch,
                                                                pooled);
  // 3. final linear -> d_out [512,128] fp32
  gin_final_kernel<<<GIN_G / 16, 256, 0, stream>>>(pooled, wfrag + 24576, linb,
                                                   (float*)d_out);
}
